// AutoCorrelation_23716809408595
// MI455X (gfx1250) — matrix operand order repaired and checked by reading, not compiled, not run on hardware
//
#include <hip/hip_runtime.h>
#include <hip/hip_bf16.h>

typedef __attribute__((ext_vector_type(16))) _Float16 v16h;
typedef __attribute__((ext_vector_type(8)))  _Float16 v8h;
typedef __attribute__((ext_vector_type(4)))  _Float16 v4h;
typedef __attribute__((ext_vector_type(8)))  float    v8f;

#define D_MODEL 1024
#define SEQ_L   2048
#define BATCH   4
#define NHEADS  16
#define DKH     64

static __device__ __forceinline__ v8f wmma_f16(v16h a, v16h b, v8f c) {
  // (neg_a, A, neg_b, B, c_mod, C, reuse_a, reuse_b)
  return __builtin_amdgcn_wmma_f32_16x16x32_f16(false, a, false, b, (short)0, c,
                                                false, false);
}

// CDNA5 async global->LDS copy (ASYNCcnt tracked, bypasses VGPRs).
// dst: workgroup-relative LDS byte offset (low 32 bits of a generic LDS ptr),
// src: 64-bit global address in a VGPR pair.
static __device__ __forceinline__ void async_load_b128(unsigned lds_off,
                                                       const void* gaddr) {
  asm volatile("global_load_async_to_lds_b128 %0, %1, off"
               :
               : "v"(lds_off), "v"(gaddr)
               : "memory");
}
static __device__ __forceinline__ void wait_async0() {
  asm volatile("s_wait_asynccnt 0x0" ::: "memory");
}

// ---------------------------------------------------------------------------
// GEMM: Out[m,n] = sum_k X[m,k] * W[n,k] + bias[n]
// Block tile 128x128, BK=32, 256 threads = 8 waves (2x4), wave tile 64x32.
// (fp32 inputs must be converted to f16, so tiles are staged via VGPRs.)
// ---------------------------------------------------------------------------
template <bool IN_F32, bool OUT_F32>
__global__ __launch_bounds__(256) void gemm_xwt_bias(
    const void* __restrict__ Xv, const float* __restrict__ W,
    const float* __restrict__ bias, void* __restrict__ Outv,
    int M, int N, int K) {
  __shared__ alignas(16) _Float16 Xs[128][40];
  __shared__ alignas(16) _Float16 Ws[128][40];

  const int tid  = threadIdx.x;
  const int lane = tid & 31;
  const int wid  = tid >> 5;
  const int wm   = wid >> 2;          // 0..1
  const int wn   = wid & 3;           // 0..3
  const int lm   = lane & 15;
  const int lg   = lane >> 4;         // half-wave group 0/1
  const int m0   = blockIdx.x * 128;
  const int n0   = blockIdx.y * 128;
  const int nk   = K >> 5;

  const float*    Xf = (const float*)Xv;
  const _Float16* Xh = (const _Float16*)Xv;

  float4 xs4[4], ws4[4];
  uint2  xs2[4];

  auto gload = [&](int kt) {
    const int kb = kt << 5;
#pragma unroll
    for (int i = 0; i < 4; ++i) {
      int id = tid + (i << 8);
      int r  = id >> 3;
      int c  = (id & 7) << 2;
      if constexpr (IN_F32) {
        xs4[i] = *(const float4*)(Xf + (size_t)(m0 + r) * K + kb + c);
      } else {
        xs2[i] = *(const uint2*)(Xh + (size_t)(m0 + r) * K + kb + c);
      }
      ws4[i] = *(const float4*)(W + (size_t)(n0 + r) * K + kb + c);
    }
  };

  auto lstore = [&]() {
#pragma unroll
    for (int i = 0; i < 4; ++i) {
      int id = tid + (i << 8);
      int r  = id >> 3;
      int c  = (id & 7) << 2;
      if constexpr (IN_F32) {
        v4h hx = {(_Float16)xs4[i].x, (_Float16)xs4[i].y,
                  (_Float16)xs4[i].z, (_Float16)xs4[i].w};
        *(v4h*)&Xs[r][c] = hx;
      } else {
        *(unsigned int*)&Xs[r][c]     = xs2[i].x;
        *(unsigned int*)&Xs[r][c + 2] = xs2[i].y;
      }
      v4h hw = {(_Float16)ws4[i].x, (_Float16)ws4[i].y,
                (_Float16)ws4[i].z, (_Float16)ws4[i].w};
      *(v4h*)&Ws[r][c] = hw;
    }
  };

  v8f acc[4][2] = {};

  gload(0);
  lstore();
  __syncthreads();

  for (int kt = 0; kt < nk; ++kt) {
    if (kt + 1 < nk) gload(kt + 1);

    // A fragments: 4 m-subtiles; B fragments: 2 n-subtiles (ISA 7.12.2 layouts)
    v16h af[4], bf[2];
    const int kb8  = lg << 3;   // 0 / 8
    const int kb16 = lg << 4;   // 0 / 16
#pragma unroll
    for (int ms = 0; ms < 4; ++ms) {
      int r = wm * 64 + ms * 16 + lm;
      ((v8h*)&af[ms])[0] = *(const v8h*)&Xs[r][kb8];
      ((v8h*)&af[ms])[1] = *(const v8h*)&Xs[r][kb8 + 16];
    }
#pragma unroll
    for (int ns = 0; ns < 2; ++ns) {
      int cI = wn * 32 + ns * 16 + lm;
      ((v8h*)&bf[ns])[0] = *(const v8h*)&Ws[cI][kb8];
      ((v8h*)&bf[ns])[1] = *(const v8h*)&Ws[cI][kb8 + 16];
    }
#pragma unroll
    for (int ms = 0; ms < 4; ++ms)
#pragma unroll
      for (int ns = 0; ns < 2; ++ns)
        acc[ms][ns] = wmma_f16(af[ms], bf[ns], acc[ms][ns]);

    __syncthreads();
    if (kt + 1 < nk) {
      lstore();
      __syncthreads();
    }
  }

  // Epilogue: C layout row = v + 8*lg, col = lm within 16x16 tile.
#pragma unroll
  for (int ms = 0; ms < 4; ++ms)
#pragma unroll
    for (int ns = 0; ns < 2; ++ns)
#pragma unroll
      for (int v = 0; v < 8; ++v) {
        int rr = m0 + wm * 64 + ms * 16 + (lg << 3) + v;
        int cc = n0 + wn * 32 + ns * 16 + lm;
        float val = acc[ms][ns][v] + bias[cc];
        if constexpr (OUT_F32)
          ((float*)Outv)[(size_t)rr * N + cc] = val;
        else
          ((_Float16*)Outv)[(size_t)rr * N + cc] = (_Float16)val;
      }
}

// ---------------------------------------------------------------------------
// Fused flash attention: one block per (b, h, 128-row q tile); 8 waves,
// each wave owns 16 q rows. K/V streamed in 64-row tiles through LDS.
// K tile uses CDNA5 async global->LDS DMA; V is transposed through VGPRs.
// ---------------------------------------------------------------------------
__global__ __launch_bounds__(256) void attn_fused(
    const _Float16* __restrict__ Q, const _Float16* __restrict__ K,
    const _Float16* __restrict__ V, _Float16* __restrict__ O) {
  __shared__ alignas(16) _Float16 Ks[64][72];      // [k_local][d]
  __shared__ alignas(16) _Float16 Vs[64][72];      // [d][k_local] (transposed)
  __shared__ alignas(16) _Float16 Ps[8][16][72];   // per-wave P tile 16x64

  const int tid  = threadIdx.x;
  const int lane = tid & 31;
  const int w    = tid >> 5;
  const int lm   = lane & 15;
  const int lg   = lane >> 4;
  const int bh   = blockIdx.y;
  const int b    = bh >> 4;
  const int h    = bh & 15;
  const int q0   = blockIdx.x * 128;

  // Q fragments for this wave's 16 rows (k = 64 -> two k-steps of 32)
  const int qr = q0 + w * 16 + lm;
  const _Float16* Qp = Q + ((size_t)(b * SEQ_L + qr) * D_MODEL + h * DKH);
  v16h qf[2];
#pragma unroll
  for (int j = 0; j < 2; ++j) {
    ((v8h*)&qf[j])[0] = *(const v8h*)(Qp + j * 32 + lg * 8);
    ((v8h*)&qf[j])[1] = *(const v8h*)(Qp + j * 32 + lg * 8 + 16);
  }

  v8f   oacc[4] = {};
  float mrow[8], lrow[8];
#pragma unroll
  for (int v = 0; v < 8; ++v) {
    mrow[v] = -1e30f;
    lrow[v] = 0.f;
  }

  const int nkt = SEQ_L / 64;
  for (int kt = 0; kt < nkt; ++kt) {
    const int kbase = kt * 64;

    // --- K tile: async DMA straight into LDS, natural [k][d] layout --------
#pragma unroll
    for (int i = 0; i < 2; ++i) {
      int id = tid + (i << 8);
      int r  = id >> 3;
      int c  = (id & 7) << 3;            // f16 units, 16B granules
      async_load_b128(
          (unsigned)(size_t)&Ks[r][c],
          (const void*)(K + ((size_t)(b * SEQ_L + kbase + r) * D_MODEL +
                             h * DKH + c)));
    }

    // --- V tile: transpose to Vs[d][k]; two k-rows packed per b32 store ----
#pragma unroll
    for (int i = 0; i < 2; ++i) {
      int id = tid + (i << 8);           // 512 ids
      int rp = (id >> 4) << 1;           // even k row
      int cb = (id & 15) << 2;           // d base (4 wide)
      const _Float16* vp =
          V + ((size_t)(b * SEQ_L + kbase + rp) * D_MODEL + h * DKH + cb);
      uint2 r0 = *(const uint2*)vp;
      uint2 r1 = *(const uint2*)(vp + D_MODEL);
      const unsigned short* e0 = (const unsigned short*)&r0;
      const unsigned short* e1 = (const unsigned short*)&r1;
#pragma unroll
      for (int j = 0; j < 4; ++j) {
        unsigned pk = (unsigned)e0[j] | ((unsigned)e1[j] << 16);
        *(unsigned*)&Vs[cb + j][rp] = pk;
      }
    }

    // Prefetch next tile (global_prefetch_b8) while this one is in flight.
    if (kt + 1 < nkt) {
      const size_t nrow = (size_t)(b * SEQ_L + kbase + 64 + (tid >> 2)) * D_MODEL +
                          h * DKH + (tid & 3) * 16;
      __builtin_prefetch(K + nrow, 0, 0);
      __builtin_prefetch(V + nrow, 0, 0);
    }

    wait_async0();       // ASYNCcnt == 0: K tile landed in LDS
    __syncthreads();

    // S = 0.125 * Q K^T + time_bias   (16 x 64 per wave, 4 n-subtiles)
    float pv[4][8];
#pragma unroll
    for (int nt = 0; nt < 4; ++nt) {
      v16h bf0, bf1;
      int krow = nt * 16 + lm;
      ((v8h*)&bf0)[0] = *(const v8h*)&Ks[krow][lg * 8];
      ((v8h*)&bf0)[1] = *(const v8h*)&Ks[krow][lg * 8 + 16];
      ((v8h*)&bf1)[0] = *(const v8h*)&Ks[krow][32 + lg * 8];
      ((v8h*)&bf1)[1] = *(const v8h*)&Ks[krow][32 + lg * 8 + 16];
      v8f s = {};
      s = wmma_f16(qf[0], bf0, s);
      s = wmma_f16(qf[1], bf1, s);
#pragma unroll
      for (int v = 0; v < 8; ++v) {
        int qg = q0 + w * 16 + (lg << 3) + v;
        int kg = kbase + nt * 16 + lm;
        pv[nt][v] = s[v] * 0.125f - 0.1f * fabsf((float)(qg - kg));
      }
    }

    // Online softmax (rows live across the 16-lane half-wave groups)
#pragma unroll
    for (int v = 0; v < 8; ++v) {
      float tm = fmaxf(fmaxf(pv[0][v], pv[1][v]), fmaxf(pv[2][v], pv[3][v]));
#pragma unroll
      for (int off = 1; off < 16; off <<= 1)
        tm = fmaxf(tm, __shfl_xor(tm, off, 32));
      float mnew  = fmaxf(mrow[v], tm);
      float alpha = __expf(mrow[v] - mnew);
      mrow[v] = mnew;
      float rs = 0.f;
#pragma unroll
      for (int nt = 0; nt < 4; ++nt) {
        float p = __expf(pv[nt][v] - mnew);
        pv[nt][v] = p;
        rs += p;
      }
#pragma unroll
      for (int off = 1; off < 16; off <<= 1) rs += __shfl_xor(rs, off, 32);
      lrow[v] = lrow[v] * alpha + rs;
#pragma unroll
      for (int dt = 0; dt < 4; ++dt) oacc[dt][v] *= alpha;
    }

    // P (C layout) -> per-wave LDS -> A fragments
#pragma unroll
    for (int nt = 0; nt < 4; ++nt)
#pragma unroll
      for (int v = 0; v < 8; ++v)
        Ps[w][(lg << 3) + v][nt * 16 + lm] = (_Float16)pv[nt][v];

    v16h pf[2];
#pragma unroll
    for (int kk = 0; kk < 2; ++kk) {
      ((v8h*)&pf[kk])[0] = *(const v8h*)&Ps[w][lm][kk * 32 + lg * 8];
      ((v8h*)&pf[kk])[1] = *(const v8h*)&Ps[w][lm][kk * 32 + lg * 8 + 16];
    }

    // O += P V  (4 d-subtiles, k = 64 -> two k-steps)
#pragma unroll
    for (int dt = 0; dt < 4; ++dt) {
      v16h vf0, vf1;
      int dcol = dt * 16 + lm;
      ((v8h*)&vf0)[0] = *(const v8h*)&Vs[dcol][lg * 8];
      ((v8h*)&vf0)[1] = *(const v8h*)&Vs[dcol][lg * 8 + 16];
      ((v8h*)&vf1)[0] = *(const v8h*)&Vs[dcol][32 + lg * 8];
      ((v8h*)&vf1)[1] = *(const v8h*)&Vs[dcol][32 + lg * 8 + 16];
      oacc[dt] = wmma_f16(pf[0], vf0, oacc[dt]);
      oacc[dt] = wmma_f16(pf[1], vf1, oacc[dt]);
    }
    __syncthreads();
  }

  // Epilogue: normalize and store f16 attention output
#pragma unroll
  for (int dt = 0; dt < 4; ++dt)
#pragma unroll
    for (int v = 0; v < 8; ++v) {
      int qg = q0 + w * 16 + (lg << 3) + v;
      int cc = h * DKH + dt * 16 + lm;
      O[(size_t)(b * SEQ_L + qg) * D_MODEL + cc] =
          (_Float16)(oacc[dt][v] / lrow[v]);
    }
}

// ---------------------------------------------------------------------------
extern "C" void kernel_launch(void* const* d_in, const int* in_sizes, int n_in,
                              void* d_out, int out_size, void* d_ws,
                              size_t ws_size, hipStream_t stream) {
  const float* queries = (const float*)d_in[0];
  const float* keys    = (const float*)d_in[1];
  const float* values  = (const float*)d_in[2];
  const float* Wq = (const float*)d_in[3];
  const float* bq = (const float*)d_in[4];
  const float* Wk = (const float*)d_in[5];
  const float* bk = (const float*)d_in[6];
  const float* Wv = (const float*)d_in[7];
  const float* bv = (const float*)d_in[8];
  const float* Wo = (const float*)d_in[9];
  const float* bo = (const float*)d_in[10];

  const int    M     = BATCH * SEQ_L;           // 8192
  const size_t elems = (size_t)M * D_MODEL;     // 8.4M f16 per buffer

  _Float16* Qb = (_Float16*)d_ws;
  _Float16* Kb = Qb + elems;
  _Float16* Vb = Kb + elems;
  _Float16* Ob = Vb + elems;

  dim3 gproj(M / 128, D_MODEL / 128);
  gemm_xwt_bias<true, false><<<gproj, 256, 0, stream>>>(queries, Wq, bq, Qb,
                                                        M, D_MODEL, D_MODEL);
  gemm_xwt_bias<true, false><<<gproj, 256, 0, stream>>>(keys, Wk, bk, Kb,
                                                        M, D_MODEL, D_MODEL);
  gemm_xwt_bias<true, false><<<gproj, 256, 0, stream>>>(values, Wv, bv, Vb,
                                                        M, D_MODEL, D_MODEL);

  dim3 gattn(SEQ_L / 128, BATCH * NHEADS);
  attn_fused<<<gattn, 256, 0, stream>>>(Qb, Kb, Vb, Ob);

  gemm_xwt_bias<false, true><<<gproj, 256, 0, stream>>>(Ob, Wo, bo,
                                                        (float*)d_out,
                                                        M, D_MODEL, D_MODEL);
}
